// SparseMatrixEmbedNet_10952166605484
// MI455X (gfx1250) — matrix operand order repaired and checked by reading, not compiled, not run on hardware
//
#include <hip/hip_runtime.h>
#include <hip/hip_fp16.h>

// ---------------------------------------------------------------------------
// CDNA5 (gfx1250) sparse-CNN embed net.
//  - f16 feature maps (halve HBM traffic; feed WMMA directly)
//  - conv layers with Cin>=32 run as implicit GEMM on v_wmma_f32_16x16x32_f16
//    (M = 16 output sites, K = Cin per 3x3 tap, N = Cout block of 16)
//  - A fragments = 2x ds_load_b128; B fragments = 2x global b128 (pre-packed)
//  - tile staging global->LDS via gfx1250 global_load_async_to_lds_b128
//    (ASYNCcnt) when available, else sync b128 fallback
//  - per-layer masked global-average pool fused into conv epilogue (atomics)
// ---------------------------------------------------------------------------

typedef __attribute__((ext_vector_type(16))) _Float16 v16h;
typedef __attribute__((ext_vector_type(8)))  float    v8f;
typedef __attribute__((ext_vector_type(4)))  int      vi4;

#if __has_builtin(__builtin_amdgcn_global_load_async_to_lds_b128) && \
    __has_builtin(__builtin_amdgcn_s_wait_asynccnt)
#define USE_ASYNC_LDS 1
typedef __attribute__((address_space(1))) vi4 gvi4;   // global (HIP __device__)
typedef __attribute__((address_space(3))) vi4 lvi4;   // LDS
#else
#define USE_ASYNC_LDS 0
#endif

// ---------------- workspace layout (bytes) ----------------
static constexpr size_t FEATA_OFF = 0;                                  // f16 [2,512,512,128] max
static constexpr size_t FEATA_SZ  = (size_t)2 * 512 * 512 * 128 * 2;    // 134,217,728
static constexpr size_t FEATB_OFF = FEATA_OFF + FEATA_SZ;               // f16 [2,512,512,32] max
static constexpr size_t FEATB_SZ  = (size_t)2 * 512 * 512 * 32 * 2;     // 33,554,432
static constexpr size_t WPACK_OFF = FEATB_OFF + FEATB_SZ;               // packed f16 weights
static constexpr size_t WPACK_HALVES = 774144;
static constexpr size_t WPACK_SZ  = WPACK_HALVES * 2;
static constexpr size_t MASK_OFF  = WPACK_OFF + WPACK_SZ;               // f32 mask pyramid
static constexpr size_t MASK_FLOATS = 699008;
static constexpr size_t POOL_OFF  = MASK_OFF + MASK_FLOATS * 4;         // f32 [2,1088] sums
static constexpr size_t CNT_OFF   = POOL_OFF + 2 * 1088 * 4;            // f32 [2,7] counts

// per-layer packed-weight offsets (halves), layers L2..L13
static constexpr size_t WOFF[12] = {
    0, 36864, 46080, 55296, 92160, 239616,
    387072, 423936, 433152, 442368, 479232, 626688};
// mask pyramid offsets (floats) per resolution 512,256,128,64,32,16,8
static constexpr size_t MOFF[7] = {0, 524288, 655360, 688128, 696320, 698368, 698880};
static constexpr int    MRES[7] = {512, 256, 128, 64, 32, 16, 8};

// ---------------- small utility kernels ----------------
__global__ void zero_f32(float* p, int n) {
  int i = blockIdx.x * 256 + threadIdx.x;
  if (i < n) p[i] = 0.f;
}

__global__ void mask_init(const unsigned char* __restrict__ mraw,
                          float* __restrict__ m512, int n) {
  int i = blockIdx.x * 256 + threadIdx.x;
  if (i < n) m512[i] = mraw[i] ? 1.f : 0.f;
}

// 2x2 max-pool downsample of mask (square maps, OH==OW)
__global__ void mask_down(const float* __restrict__ min_,
                          float* __restrict__ mout, int OW, int n) {
  int i = blockIdx.x * 256 + threadIdx.x;
  if (i >= n) return;
  int x = i % OW;
  int y = (i / OW) % OW;
  int b = i / (OW * OW);
  int IW = OW * 2;
  size_t base = ((size_t)b * IW + 2 * y) * IW + 2 * x;
  float v = fmaxf(fmaxf(min_[base], min_[base + 1]),
                  fmaxf(min_[base + IW], min_[base + IW + 1]));
  mout[i] = v;
}

// count active sites per image at one resolution
__global__ void mask_count(const float* __restrict__ m, float* __restrict__ cnt,
                           int nPerImg, int res) {
  __shared__ float s[256];
  int tid = threadIdx.x;
  int b = blockIdx.y;
  int i = blockIdx.x * 256 + tid;
  float v = (i < nPerImg) ? m[(size_t)b * nPerImg + i] : 0.f;
  s[tid] = v;
  __syncthreads();
  for (int o = 128; o > 0; o >>= 1) {
    if (tid < o) s[tid] += s[tid + o];
    __syncthreads();
  }
  if (tid == 0) atomicAdd(&cnt[b * 7 + res], s[0]);
}

// ---------------- weight packing into WMMA B-fragment order ----------------
// B-matrix (K x 16) per-lane layout: n = lane&15, k = (lane&16) + slot.
// dst half index = frag*512 + lane*16 + slot, frag = (tap*KC + kc)*NB + nb.
__global__ void pack_weights(const float* __restrict__ w, _Float16* __restrict__ dst,
                             int CIN, int COUT, int total) {
  int idx = blockIdx.x * 256 + threadIdx.x;
  if (idx >= total) return;
  int slot = idx & 15;
  int lane = (idx >> 4) & 31;
  int frag = idx >> 9;
  int NB = COUT >> 4, KC = CIN >> 5;
  int nb = frag % NB;
  int rest = frag / NB;
  int kc = rest % KC;
  int t = rest / KC;                       // tap 0..8 (ky*3+kx)
  int k = kc * 32 + (lane & 16) + slot;    // input channel
  int n = nb * 16 + (lane & 15);           // output channel
  dst[idx] = (_Float16)w[((size_t)t * CIN + k) * COUT + n];
}

// ---------------- layer 1: 5x5 conv, Cin=1 (VALU; compute-light) ----------
__launch_bounds__(256)
__global__ void conv1_5x5(const float* __restrict__ x, const float* __restrict__ m512,
                          const float* __restrict__ k1, _Float16* __restrict__ fout,
                          float* __restrict__ pool) {
  __shared__ float w[25 * 128];
  __shared__ float ps[128];
  int tid = threadIdx.x;
  for (int i = tid; i < 25 * 128; i += 256) w[i] = k1[i];
  if (tid < 128) ps[tid] = 0.f;
  __syncthreads();

  int p = blockIdx.x * 256 + tid;      // [b][y][x], 2^18 pixels per image
  int b = p >> 18;
  int y = (p >> 9) & 511;
  int xq = p & 511;
  float xv[25];
#pragma unroll
  for (int t = 0; t < 25; ++t) {
    int yy = y + t / 5 - 2, xx = xq + t % 5 - 2;
    xv[t] = ((unsigned)yy < 512u && (unsigned)xx < 512u)
                ? x[((size_t)b << 18) + yy * 512 + xx]
                : 0.f;
  }
  float mk = m512[p];
  _Float16* outp = fout + (size_t)p * 128;
  for (int c = 0; c < 128; ++c) {
    float a = 0.f;
#pragma unroll
    for (int t = 0; t < 25; ++t) a = fmaf(xv[t], w[t * 128 + c], a);
    float v = fmaxf(a, 0.f) * mk;
    outp[c] = (_Float16)v;
    atomicAdd(&ps[c], v);
  }
  __syncthreads();
  if (tid < 128) atomicAdd(&pool[b * 1088 + tid], ps[tid]);
}

// ---------------- WMMA conv (3x3, SAME, fused relu*mask + pool) -----------
// Block = 128 threads (4 waves). Each block computes an 8x8 output tile; each
// wave owns 2 output rows (16 sites = WMMA M). K loop: 9 taps x (CIN/32)
// chunks of v_wmma_f32_16x16x32_f16; N loop: COUT/16 blocks.
template <int CIN, int COUT, int STRIDE, int DIL>
__launch_bounds__(128)
__global__ void sconv_wmma(const _Float16* __restrict__ fin, _Float16* __restrict__ fout,
                           const float* __restrict__ mout, const _Float16* __restrict__ wpk,
                           float* __restrict__ pool, int OH, int OW, int poolOff) {
  constexpr int SPAN = 7 * STRIDE + 2 * DIL + 1;     // input tile incl. halo
  constexpr int PAD = (STRIDE == 1) ? DIL : 0;       // SAME: pad_lo (stride2 -> 0)
  constexpr int KC = CIN / 32, NB = COUT / 16;
  extern __shared__ _Float16 tile[];                 // [SPAN][SPAN][CIN]

  const int b = blockIdx.z;
  const int oy0 = blockIdx.y * 8, ox0 = blockIdx.x * 8;
  const int INH = OH * STRIDE, INW = OW * STRIDE;
  const int iy0 = oy0 * STRIDE - PAD, ix0 = ox0 * STRIDE - PAD;
  const int tid = threadIdx.x;

  // cooperative tile staging in 16-byte quanta; zero-fill OOB (SAME padding)
  constexpr int QUADS = SPAN * SPAN * CIN / 8;       // 8 halves per transfer
  uint4* tq = (uint4*)tile;
  for (int i = tid; i < QUADS; i += 128) {
    int c8 = i % (CIN / 8);
    int pp = i / (CIN / 8);
    int ix = pp % SPAN, iy = pp / SPAN;
    int gy = iy0 + iy, gx = ix0 + ix;
    bool inb = ((unsigned)gy < (unsigned)INH) && ((unsigned)gx < (unsigned)INW);
    const _Float16* gsrc = fin + (((size_t)b * INH + gy) * INW + gx) * CIN + 8 * c8;
#if USE_ASYNC_LDS
    if (inb) {
      __builtin_amdgcn_global_load_async_to_lds_b128(
          (gvi4*)gsrc, (lvi4*)(tile + 8 * (size_t)i), 0, 0);
    } else {
      uint4 z = {0u, 0u, 0u, 0u};
      tq[i] = z;
    }
#else
    uint4 v = {0u, 0u, 0u, 0u};
    if (inb) v = *(const uint4*)gsrc;
    tq[i] = v;
#endif
  }
#if USE_ASYNC_LDS
  __builtin_amdgcn_s_wait_asynccnt(0);
#endif
  __syncthreads();

  const int wave = tid >> 5, lane = tid & 31;
  // A fragment: row m = lane&15 -> local (row, col); K base 0 / 16 per half-wave
  const int mA = lane & 15;
  const int ayl = 2 * wave + (mA >> 3);
  const int axl = mA & 7;
  const int kbase = (lane < 16) ? 0 : 16;

  v8f acc[NB];
  v8f vzero = {};
#pragma unroll
  for (int nb = 0; nb < NB; ++nb) acc[nb] = vzero;

  union AF { v16h h; uint4 q[2]; };
  union BF { v16h h; uint4 q[2]; };

  for (int t = 0; t < 9; ++t) {
    const int ky = t / 3, kx = t % 3;
    const int iy = ayl * STRIDE + ky * DIL;
    const int ix = axl * STRIDE + kx * DIL;
    const int base = (iy * SPAN + ix) * CIN;
#pragma unroll
    for (int kc = 0; kc < KC; ++kc) {
      // A fragment: 2x ds_load_b128 (VGPRs 0..3 = k kbase..kbase+7,
      // VGPRs 4..7 = k kbase+8..kbase+15; one contiguous run in LDS)
      AF a;
      const _Float16* asrc = tile + base + kc * 32 + kbase;
      a.q[0] = *(const uint4*)(asrc);
      a.q[1] = *(const uint4*)(asrc + 8);
      const uint4* wq =
          (const uint4*)wpk + ((((size_t)(t * KC + kc)) * NB * 512 + (size_t)lane * 16) >> 3);
      __builtin_prefetch((const void*)(wq + NB * 64), 0, 0);  // next fragment group
#pragma unroll
      for (int nb = 0; nb < NB; ++nb) {
        BF bf;
        bf.q[0] = wq[nb * 64];
        bf.q[1] = wq[nb * 64 + 1];
        acc[nb] = __builtin_amdgcn_wmma_f32_16x16x32_f16(
            false, a.h, false, bf.h, (short)0, acc[nb], false, false);
      }
    }
  }

  // D layout: vgpr r, lanes<16 -> (M=r, N=lane); lanes>=16 -> (M=r+8, N=lane-16)
  const int oy = oy0 + 2 * wave + (lane >> 4);
  float mk[8];
#pragma unroll
  for (int r = 0; r < 8; ++r)
    mk[r] = mout[((size_t)b * OH + oy) * OW + (ox0 + r)];

#pragma unroll
  for (int nb = 0; nb < NB; ++nb) {
    int co = nb * 16 + (lane & 15);
    float psum = 0.f;
#pragma unroll
    for (int r = 0; r < 8; ++r) {
      int ox = ox0 + r;
      float v = fmaxf(acc[nb][r], 0.f) * mk[r];
      psum += v;
      fout[(((size_t)b * OH + oy) * OW + ox) * COUT + co] = (_Float16)v;
    }
    atomicAdd(&pool[b * 1088 + poolOff + co], psum);
  }
}

// ---------------- MLP head: 1088 -> 512 (relu) -> 128 ----------------------
__launch_bounds__(512)
__global__ void mlp_head(const float* __restrict__ pool, const float* __restrict__ cnt,
                         const float* __restrict__ fw1, const float* __restrict__ fb1,
                         const float* __restrict__ fw2, const float* __restrict__ fb2,
                         float* __restrict__ out) {
  __shared__ float pn[1088];
  __shared__ float h1[512];
  int b = blockIdx.x, tid = threadIdx.x;
  for (int c = tid; c < 1088; c += 512) {
    int res;  // channel -> output resolution of its layer
    if (c < 224) res = 0;         // L1..L4 @512
    else if (c < 352) res = 1;    // L5 @256
    else if (c < 480) res = 2;    // L6 @128
    else if (c < 704) res = 3;    // L7..L10 @64
    else if (c < 832) res = 4;    // L11 @32
    else if (c < 960) res = 5;    // L12 @16
    else res = 6;                 // L13 @8
    pn[c] = pool[b * 1088 + c] / fmaxf(cnt[b * 7 + res], 1.f);
  }
  __syncthreads();
  float s = fb1[tid];
  for (int c = 0; c < 1088; ++c) s = fmaf(pn[c], fw1[c * 512 + tid], s);
  h1[tid] = fmaxf(s, 0.f);
  __syncthreads();
  if (tid < 128) {
    float o = fb2[tid];
    for (int c = 0; c < 512; ++c) o = fmaf(h1[c], fw2[c * 128 + tid], o);
    out[b * 128 + tid] = o;
  }
}

// ---------------------------------------------------------------------------
extern "C" void kernel_launch(void* const* d_in, const int* in_sizes, int n_in,
                              void* d_out, int out_size, void* d_ws, size_t ws_size,
                              hipStream_t stream) {
  const float* x = (const float*)d_in[0];
  const unsigned char* mraw = (const unsigned char*)d_in[1];  // bernoulli bool mask
  const float* kk[13];
  for (int i = 0; i < 13; ++i) kk[i] = (const float*)d_in[2 + i];
  const float* fw1 = (const float*)d_in[15];
  const float* fb1 = (const float*)d_in[16];
  const float* fw2 = (const float*)d_in[17];
  const float* fb2 = (const float*)d_in[18];
  float* out = (float*)d_out;

  char* ws = (char*)d_ws;
  _Float16* featA = (_Float16*)(ws + FEATA_OFF);
  _Float16* featB = (_Float16*)(ws + FEATB_OFF);
  _Float16* wpk = (_Float16*)(ws + WPACK_OFF);
  float* mbase = (float*)(ws + MASK_OFF);
  float* pool = (float*)(ws + POOL_OFF);
  float* cnt = (float*)(ws + CNT_OFF);
  float* m[7];
  for (int r = 0; r < 7; ++r) m[r] = mbase + MOFF[r];

  // 0) zero pooled sums + counts (pool and cnt are contiguous)
  zero_f32<<<(2190 + 255) / 256, 256, 0, stream>>>(pool, 2176 + 14);

  // 1) mask pyramid + active counts
  mask_init<<<(2 * 512 * 512) / 256, 256, 0, stream>>>(mraw, m[0], 2 * 512 * 512);
  for (int r = 1; r < 7; ++r) {
    int OW = MRES[r];
    int n = 2 * OW * OW;
    mask_down<<<(n + 255) / 256, 256, 0, stream>>>(m[r - 1], m[r], OW, n);
  }
  for (int r = 0; r < 7; ++r) {
    int np = MRES[r] * MRES[r];
    mask_count<<<dim3((np + 255) / 256, 2), 256, 0, stream>>>(m[r], cnt, np, r);
  }

  // 2) pack conv weights (L2..L13) into WMMA B-fragment order, f32 -> f16
  static const int PCIN[12] = {128, 32, 32, 32, 128, 128, 128, 32, 32, 32, 128, 128};
  static const int PCOUT[12] = {32, 32, 32, 128, 128, 128, 32, 32, 32, 128, 128, 128};
  for (int l = 0; l < 12; ++l) {
    int total = 9 * PCIN[l] * PCOUT[l];
    pack_weights<<<(total + 255) / 256, 256, 0, stream>>>(kk[l + 1], wpk + WOFF[l],
                                                          PCIN[l], PCOUT[l], total);
  }

  // 3) layer 1 (Cin=1, 5x5) -> featA
  conv1_5x5<<<2048, 256, 0, stream>>>(x, m[0], kk[0], featA, pool);

  // 4) WMMA conv layers (ping-pong featA/featB); shmem = SPAN^2 * CIN * 2 bytes
  sconv_wmma<128, 32, 1, 1><<<dim3(64, 64, 2), 128, 10 * 10 * 128 * 2, stream>>>(
      featA, featB, m[0], wpk + WOFF[0], pool, 512, 512, 128);                    // L2
  sconv_wmma<32, 32, 1, 2><<<dim3(64, 64, 2), 128, 12 * 12 * 32 * 2, stream>>>(
      featB, featA, m[0], wpk + WOFF[1], pool, 512, 512, 160);                    // L3
  sconv_wmma<32, 32, 1, 3><<<dim3(64, 64, 2), 128, 14 * 14 * 32 * 2, stream>>>(
      featA, featB, m[0], wpk + WOFF[2], pool, 512, 512, 192);                    // L4
  sconv_wmma<32, 128, 2, 1><<<dim3(32, 32, 2), 128, 17 * 17 * 32 * 2, stream>>>(
      featB, featA, m[1], wpk + WOFF[3], pool, 256, 256, 224);                    // L5
  sconv_wmma<128, 128, 2, 1><<<dim3(16, 16, 2), 128, 17 * 17 * 128 * 2, stream>>>(
      featA, featB, m[2], wpk + WOFF[4], pool, 128, 128, 352);                    // L6
  sconv_wmma<128, 128, 2, 1><<<dim3(8, 8, 2), 128, 17 * 17 * 128 * 2, stream>>>(
      featB, featA, m[3], wpk + WOFF[5], pool, 64, 64, 480);                      // L7
  sconv_wmma<128, 32, 1, 1><<<dim3(8, 8, 2), 128, 10 * 10 * 128 * 2, stream>>>(
      featA, featB, m[3], wpk + WOFF[6], pool, 64, 64, 608);                      // L8
  sconv_wmma<32, 32, 1, 2><<<dim3(8, 8, 2), 128, 12 * 12 * 32 * 2, stream>>>(
      featB, featA, m[3], wpk + WOFF[7], pool, 64, 64, 640);                      // L9
  sconv_wmma<32, 32, 1, 3><<<dim3(8, 8, 2), 128, 14 * 14 * 32 * 2, stream>>>(
      featA, featB, m[3], wpk + WOFF[8], pool, 64, 64, 672);                      // L10
  sconv_wmma<32, 128, 2, 1><<<dim3(4, 4, 2), 128, 17 * 17 * 32 * 2, stream>>>(
      featB, featA, m[4], wpk + WOFF[9], pool, 32, 32, 704);                      // L11
  sconv_wmma<128, 128, 2, 1><<<dim3(2, 2, 2), 128, 17 * 17 * 128 * 2, stream>>>(
      featA, featB, m[5], wpk + WOFF[10], pool, 16, 16, 832);                     // L12
  sconv_wmma<128, 128, 2, 1><<<dim3(1, 1, 2), 128, 17 * 17 * 128 * 2, stream>>>(
      featB, featA, m[6], wpk + WOFF[11], pool, 8, 8, 960);                       // L13

  // 5) MLP head
  mlp_head<<<2, 512, 0, stream>>>(pool, cnt, fw1, fb1, fw2, fb2, out);

  (void)in_sizes; (void)n_in; (void)out_size; (void)ws_size;
}
